// ResnetAttentionBlock_42331197670011
// MI455X (gfx1250) — compile-verified, matrix operand order checked by reading, not run on hardware
//
#include <hip/hip_runtime.h>
#include <hip/hip_bf16.h>

#define GNUM 32
#define DIM 256
#define TDIM 1024
#define BATCH 8
#define HW 4096
#define EPS 1e-5f
#define USE_ASYNC_COPY 1

typedef __attribute__((ext_vector_type(16))) _Float16     v16h;
typedef __attribute__((ext_vector_type(8)))  _Float16     v8h;
typedef __attribute__((ext_vector_type(4)))  _Float16     v4h;
typedef __attribute__((ext_vector_type(8)))  float        v8f;
typedef __attribute__((ext_vector_type(8)))  unsigned int v8u;

// ---- WMMA helpers (CDNA5: V_WMMA_F32_16X16X32_F16, wave32) ----
// 16-bit A/B element->K mapping per ISA 7.12.2:  k = (e&7) + ((e&8)<<1) + (s<<3)
// inverse: s = (k>>3)&1 ; e = (k&7) + ((k&16)>>1)   (8 consecutive k -> 8 consecutive e)
__device__ __forceinline__ int frag_k(int e, int s) {
  return (e & 7) + ((e & 8) << 1) + (s << 3);
}

__device__ __forceinline__ v8f wmma_f16(v16h a, v16h b, v8f c) {
  return __builtin_amdgcn_wmma_f32_16x16x32_f16(
      false, a, false, b, (short)0, c, false, false);
}

// fragment load from fragment-ordered LDS tile: 32B contiguous per lane -> 2x ds_load_b128
__device__ __forceinline__ v16h lds_frag(const _Float16* base, int lane) {
  v8u r = *(const v8u*)(const void*)(base + lane * 16);
  return __builtin_bit_cast(v16h, r);
}

// ---- async global->LDS copy (gfx1250 GLOBAL_LOAD_ASYNC_TO_LDS_B128, ASYNCcnt) ----
__device__ __forceinline__ void async_copy_b128(const _Float16* gsrc, _Float16* ldst) {
#if USE_ASYNC_COPY
  unsigned int loff = (unsigned int)(unsigned long long)(uintptr_t)ldst;  // LDS offset = low 32 bits
  unsigned long long ga = (unsigned long long)(uintptr_t)gsrc;
  asm volatile("global_load_async_to_lds_b128 %0, %1, off"
               :: "v"(loff), "v"(ga) : "memory");
#else
  *(v8h*)(void*)ldst = *(const v8h*)(const void*)gsrc;
#endif
}
__device__ __forceinline__ void async_wait() {
#if USE_ASYNC_COPY
  asm volatile("s_wait_asynccnt 0x0" ::: "memory");
#endif
}

// ---------------- weight packing: f32 -> f16 in fragment order ----------------
// packed[(ks*ntiles + nb)*2048 + nf*512 + lane*16 + e], k = ks*32 + frag_k(e, lane>>4),
// n = nb*64 + nf*16 + (lane&15)
__global__ __launch_bounds__(256) void pack_weights(
    const float* __restrict__ w1, const float* __restrict__ w2,
    const float* __restrict__ wqkv, const float* __restrict__ wout,
    _Float16* __restrict__ w1h, _Float16* __restrict__ w2h,
    _Float16* __restrict__ wqkvh, _Float16* __restrict__ wouth) {
  const int W9 = 9 * DIM * DIM;          // 589824 (72 k-steps * 4 ntiles * 2048)
  const int QK = DIM * 768;              // 196608 (8 k-steps * 12 ntiles * 2048)
  int i = blockIdx.x * 256 + threadIdx.x;
  int chunk = i & 2047;
  int e = chunk & 15, lane = (chunk >> 4) & 31, nf = chunk >> 9;
  int kin = frag_k(e, lane >> 4);
  int nl = nf * 16 + (lane & 15);
  if (i < W9) {
    int t = i >> 11; int nb = t & 3; int ks = t >> 2;    // ks = tap*8 + kk
    int tap = ks >> 3, kk = ks & 7;
    int c = kk * 32 + kin, n = nb * 64 + nl;
    w1h[i] = (_Float16)w1[(n * DIM + c) * 9 + tap];
  } else if (i < 2 * W9) {
    int j = i - W9; int t = j >> 11; int nb = t & 3; int ks = t >> 2;
    int tap = ks >> 3, kk = ks & 7;
    int c = kk * 32 + kin, n = nb * 64 + nl;
    w2h[j] = (_Float16)w2[(n * DIM + c) * 9 + tap];
  } else if (i < 2 * W9 + QK) {
    int j = i - 2 * W9; int t = j >> 11; int nb = t % 12; int kk = t / 12;
    int c = kk * 32 + kin, n = nb * 64 + nl;
    wqkvh[j] = (_Float16)wqkv[n * DIM + c];
  } else if (i < 2 * W9 + QK + DIM * DIM) {
    int j = i - 2 * W9 - QK; int t = j >> 11; int nb = t & 3; int kk = t >> 2;
    int c = kk * 32 + kin, n = nb * 64 + nl;
    wouth[j] = (_Float16)wout[n * DIM + c];
  }
}

// ---------------- time-embedding MLP ----------------
__global__ __launch_bounds__(256) void time_mlp(
    const float* __restrict__ te, const float* __restrict__ w,
    const float* __restrict__ bias, float* __restrict__ out) {
  int b = blockIdx.x, n = threadIdx.x;
  float acc = bias[n];
  for (int k = 0; k < TDIM; ++k) {
    float t = te[b * TDIM + k];
    acc += (t / (1.f + expf(-t))) * w[k * DIM + n];
  }
  out[b * DIM + n] = acc;
}

// ---------------- GroupNorm (+optional SiLU) -> f16, float4 vectorized ----------------
__global__ __launch_bounds__(256) void gn_silu(
    const float* __restrict__ x, const float* __restrict__ gam,
    const float* __restrict__ bet, _Float16* __restrict__ y, int do_silu) {
  const int CPG = DIM / GNUM;      // 8
  const int N4 = CPG * HW / 4;     // 8192 float4
  int b = blockIdx.x / GNUM, grp = blockIdx.x % GNUM;
  const float4* xp4 = (const float4*)(x + ((size_t)b * DIM + grp * CPG) * HW);
  float s = 0.f, q = 0.f;
  for (int i = threadIdx.x; i < N4; i += 256) {
    float4 v = xp4[i];
    s += v.x + v.y + v.z + v.w;
    q += v.x * v.x + v.y * v.y + v.z * v.z + v.w * v.w;
  }
  __shared__ float ss[256], sq[256];
  ss[threadIdx.x] = s; sq[threadIdx.x] = q; __syncthreads();
  for (int o = 128; o > 0; o >>= 1) {
    if (threadIdx.x < o) { ss[threadIdx.x] += ss[threadIdx.x + o]; sq[threadIdx.x] += sq[threadIdx.x + o]; }
    __syncthreads();
  }
  float mean = ss[0] / (float)(N4 * 4);
  float var = sq[0] / (float)(N4 * 4) - mean * mean;
  float rstd = rsqrtf(var + EPS);
  _Float16* yp = y + ((size_t)b * DIM + grp * CPG) * HW;
  for (int i = threadIdx.x; i < N4; i += 256) {
    int c = grp * CPG + (i >> 10);
    float g = gam[c] * rstd, bb = bet[c] - mean * g;
    float4 v = xp4[i];
    float o0 = v.x * g + bb, o1 = v.y * g + bb, o2 = v.z * g + bb, o3 = v.w * g + bb;
    if (do_silu) {
      o0 = o0 / (1.f + expf(-o0)); o1 = o1 / (1.f + expf(-o1));
      o2 = o2 / (1.f + expf(-o2)); o3 = o3 / (1.f + expf(-o3));
    }
    v4h hv; hv[0] = (_Float16)o0; hv[1] = (_Float16)o1; hv[2] = (_Float16)o2; hv[3] = (_Float16)o3;
    *(v4h*)(void*)(yp + i * 4) = hv;
  }
}

// ---------------- 3x3 conv as implicit GEMM (M=64 spatial row, N=64, K=2304) ----------------
__global__ __launch_bounds__(256) void conv3x3_wmma(
    const _Float16* __restrict__ inh,   // [B, 256, HW] f16
    const _Float16* __restrict__ wh,    // fragment-packed [72 ks][4 nb][2048]
    const float* __restrict__ bias,
    const float* __restrict__ addvec,   // [B, 256] or null
    const float* __restrict__ addfull,  // [B, 256, HW] or null
    float* __restrict__ out) {          // [B, 256, HW] f32
  __shared__ _Float16 a_sw[4 * 512] __attribute__((aligned(32)));  // 4 m-subtiles
  __shared__ _Float16 b_sw[4 * 512] __attribute__((aligned(32)));  // 4 n-frags
  int tile = blockIdx.x;                // B * 64 * 4
  int nt = tile & 3;
  int y0 = (tile >> 2) & 63;
  int b  = tile >> 8;
  int n_base = nt * 64;
  int tid = threadIdx.x, lane = tid & 31, w = tid >> 5;
  int mi = w & 3, ni = w >> 2;          // 4x2 wave grid -> 64M x 64N tile
  // A-staging geometry: thread -> (m, 8 consecutive kc)
  int stg_m = tid >> 2, stg_kcb = (tid & 3) * 8;
  int stg_s = (stg_kcb >> 3) & 1, stg_e0 = (stg_kcb & 16) >> 1;
  _Float16* a_dst = a_sw + (stg_m >> 4) * 512 + (((stg_m & 15) + 16 * stg_s) * 16) + stg_e0;
  v8f acc0 = {}, acc1 = {};
  for (int tap = 0; tap < 9; ++tap) {
    int dy = tap / 3 - 1, dx = tap % 3 - 1;
    int yy = y0 + dy;
    int xx = stg_m + dx;
    bool valid = (yy >= 0) & (yy < 64) & (xx >= 0) & (xx < 64);
    for (int kk = 0; kk < 8; ++kk) {
      int ks = tap * 8 + kk;
      // B tile: contiguous 4KB of fragment-packed weights -> async to LDS
      async_copy_b128(wh + ((size_t)(ks * 4 + nt) * 2048) + tid * 8, b_sw + tid * 8);
      // A tile: 8 strided channel loads, one b128 LDS store (fragment order)
      v8h hv = {};
      if (valid) {
        const _Float16* src = inh + ((size_t)b * DIM + kk * 32 + stg_kcb) * HW + yy * 64 + xx;
#pragma unroll
        for (int j = 0; j < 8; ++j) hv[j] = src[(size_t)j * HW];
      }
      *(v8h*)(void*)a_dst = hv;
      async_wait();
      __syncthreads();
      v16h a  = lds_frag(a_sw + mi * 512, lane);
      v16h b0 = lds_frag(b_sw + (2 * ni + 0) * 512, lane);
      v16h b1 = lds_frag(b_sw + (2 * ni + 1) * 512, lane);
      acc0 = wmma_f16(a, b0, acc0);
      acc1 = wmma_f16(a, b1, acc1);
      __syncthreads();
    }
  }
  int hi = lane >> 4, nl = lane & 15;
#pragma unroll
  for (int f = 0; f < 2; ++f) {
    v8f acc = f ? acc1 : acc0;
    int n = n_base + 32 * ni + 16 * f + nl;
    float bn = bias[n] + (addvec ? addvec[b * DIM + n] : 0.f);
    size_t idx = ((size_t)b * DIM + n) * HW + y0 * 64 + 16 * mi + 8 * hi;
    v8f val;
#pragma unroll
    for (int v = 0; v < 8; ++v) val[v] = acc[v] + bn;
    if (addfull) {
      v8f r = *(const v8f*)(const void*)(addfull + idx);
#pragma unroll
      for (int v = 0; v < 8; ++v) val[v] += r[v];
    }
    *(v8f*)(void*)(out + idx) = val;
  }
}

// ---------------- 1x1 conv GEMM (K=256) ----------------
__global__ __launch_bounds__(256) void gemm1x1_wmma(
    const _Float16* __restrict__ inh,   // [B, 256, HW] f16
    const _Float16* __restrict__ wh,    // fragment-packed [8 ks][ntiles][2048]
    const float* __restrict__ bias,
    const float* __restrict__ addfull,  // [B, N, HW] or null
    float* __restrict__ outf,           // or null
    _Float16* __restrict__ outh,        // or null
    int N) {
  __shared__ _Float16 a_sw[4 * 512] __attribute__((aligned(32)));
  __shared__ _Float16 b_sw[4 * 512] __attribute__((aligned(32)));
  int ntiles = N >> 6;
  int tile = blockIdx.x;
  int nt = tile % ntiles;
  int pt = (tile / ntiles) & 63;
  int b  = tile / (ntiles * 64);
  int n_base = nt * 64, p_base = pt * 64;
  int tid = threadIdx.x, lane = tid & 31, w = tid >> 5;
  int mi = w & 3, ni = w >> 2;
  int stg_m = tid >> 2, stg_kcb = (tid & 3) * 8;
  int stg_s = (stg_kcb >> 3) & 1, stg_e0 = (stg_kcb & 16) >> 1;
  _Float16* a_dst = a_sw + (stg_m >> 4) * 512 + (((stg_m & 15) + 16 * stg_s) * 16) + stg_e0;
  v8f acc0 = {}, acc1 = {};
  for (int kk = 0; kk < 8; ++kk) {
    async_copy_b128(wh + ((size_t)(kk * ntiles + nt) * 2048) + tid * 8, b_sw + tid * 8);
    const _Float16* src = inh + ((size_t)b * DIM + kk * 32 + stg_kcb) * HW + p_base + stg_m;
    v8h hv;
#pragma unroll
    for (int j = 0; j < 8; ++j) hv[j] = src[(size_t)j * HW];
    *(v8h*)(void*)a_dst = hv;
    async_wait();
    __syncthreads();
    v16h a  = lds_frag(a_sw + mi * 512, lane);
    v16h b0 = lds_frag(b_sw + (2 * ni + 0) * 512, lane);
    v16h b1 = lds_frag(b_sw + (2 * ni + 1) * 512, lane);
    acc0 = wmma_f16(a, b0, acc0);
    acc1 = wmma_f16(a, b1, acc1);
    __syncthreads();
  }
  int hi = lane >> 4, nl = lane & 15;
#pragma unroll
  for (int f = 0; f < 2; ++f) {
    v8f acc = f ? acc1 : acc0;
    int n = n_base + 32 * ni + 16 * f + nl;
    float bn = bias[n];
    size_t idx = ((size_t)b * N + n) * HW + p_base + 16 * mi + 8 * hi;
    v8f val;
#pragma unroll
    for (int v = 0; v < 8; ++v) val[v] = acc[v] + bn;
    if (addfull) {
      v8f r = *(const v8f*)(const void*)(addfull + idx);
#pragma unroll
      for (int v = 0; v < 8; ++v) val[v] += r[v];
    }
    if (outf) *(v8f*)(void*)(outf + idx) = val;
    if (outh) {
      v8h hv2;
#pragma unroll
      for (int v = 0; v < 8; ++v) hv2[v] = (_Float16)val[v];
      *(v8h*)(void*)(outh + idx) = hv2;
    }
  }
}

// ---------------- flash attention: seq=4096, C=256, single head, scale=1/16 ----------------
__global__ __launch_bounds__(256) void attn_wmma(
    const _Float16* __restrict__ qkv,  // [B, 768, HW] f16 (q:0..255, k:256..511, v:512..767)
    _Float16* __restrict__ o) {        // [B, 256, HW] f16
  __shared__ _Float16 k_sw[16 * 512] __attribute__((aligned(32)));  // 8 ksteps x 2 jfrags
  __shared__ _Float16 v_sw[16 * 512] __attribute__((aligned(32)));  // 16 n-frags
  __shared__ _Float16 p_sw[8 * 512]  __attribute__((aligned(32)));  // per-wave P (A order)
  int b  = blockIdx.x >> 5;
  int qt = blockIdx.x & 31;
  int tid = threadIdx.x, lane = tid & 31, w = tid >> 5;
  int hi = lane >> 4, nl = lane & 15;
  const _Float16* qp = qkv + (size_t)b * 768 * HW;
  int m0 = qt * 128 + 16 * w;              // this wave's 16 query rows
  // Q fragments resident in registers: 16 rows x 256 channels (8 k-steps)
  v16h qa[8];
#pragma unroll
  for (int kk = 0; kk < 8; ++kk) {
#pragma unroll
    for (int e = 0; e < 16; ++e)
      qa[kk][e] = qp[(size_t)(32 * kk + frag_k(e, hi)) * HW + m0 + nl];
  }
  v8f zero = {};
  v8f oacc[16];
#pragma unroll
  for (int f = 0; f < 16; ++f) oacc[f] = zero;
  float mrow[8], rsum[8];
#pragma unroll
  for (int v = 0; v < 8; ++v) { mrow[v] = -1e30f; rsum[v] = 0.f; }
  const float scale = 0.0625f;
  // P-store geometry (A-fragment order): col=nl -> e=nl&7, s=(nl>>3)&1; col=16+nl -> e+8
  int p_s = (nl >> 3) & 1, p_e0 = nl & 7;

  for (int j0 = 0; j0 < HW; j0 += 32) {
    // V tile: global-contiguous 16B groups -> async copy into fragment-ordered LDS
#pragma unroll
    for (int it = 0; it < 4; ++it) {
      int g = tid + 256 * it;              // 1024 groups of 8
      int c = g & 255, jg = g >> 8, jb = jg * 8;
      int s = jg & 1, e0 = (jg & 2) << 2;
      _Float16* dst = v_sw + (c >> 4) * 512 + ((c & 15) + 16 * s) * 16 + e0;
      async_copy_b128(qp + (size_t)(512 + c) * HW + j0 + jb, dst);
    }
    // K tile: 8 strided channel loads per group, one b128 LDS store
#pragma unroll
    for (int it = 0; it < 4; ++it) {
      int g = tid + 256 * it;
      int cb = (g & 31) * 8, j = g >> 5;
      int s = (cb >> 3) & 1, e0 = (cb & 16) >> 1, kk = cb >> 5;
      _Float16* dst = k_sw + (kk * 2 + (j >> 4)) * 512 + ((j & 15) + 16 * s) * 16 + e0;
      const _Float16* src = qp + (size_t)(256 + cb) * HW + j0 + j;
      v8h hv;
#pragma unroll
      for (int jj = 0; jj < 8; ++jj) hv[jj] = src[(size_t)jj * HW];
      *(v8h*)(void*)dst = hv;
    }
    if (j0 + 32 < HW)
      __builtin_prefetch(qp + (size_t)256 * HW + j0 + 32, 0, 1);
    async_wait();
    __syncthreads();

    // S = Q @ K^T  (16 x 32 per wave)
    v8f s0 = zero, s1 = zero;
#pragma unroll
    for (int kk = 0; kk < 8; ++kk) {
      v16h b0 = lds_frag(k_sw + (kk * 2 + 0) * 512, lane);
      v16h b1 = lds_frag(k_sw + (kk * 2 + 1) * 512, lane);
      s0 = wmma_f16(qa[kk], b0, s0);
      s1 = wmma_f16(qa[kk], b1, s1);
    }
#pragma unroll
    for (int v = 0; v < 8; ++v) { s0[v] *= scale; s1[v] *= scale; }

    // online softmax per row
#pragma unroll
    for (int v = 0; v < 8; ++v) {
      float m2 = fmaxf(s0[v], s1[v]);
#pragma unroll
      for (int d = 1; d < 16; d <<= 1) m2 = fmaxf(m2, __shfl_xor(m2, d, 32));
      float nm = fmaxf(mrow[v], m2);
      float corr = __expf(mrow[v] - nm);
      mrow[v] = nm;
      float p0 = __expf(s0[v] - nm), p1 = __expf(s1[v] - nm);
      float psum = p0 + p1;
#pragma unroll
      for (int d = 1; d < 16; d <<= 1) psum += __shfl_xor(psum, d, 32);
      rsum[v] = rsum[v] * corr + psum;
#pragma unroll
      for (int f = 0; f < 16; ++f) oacc[f][v] *= corr;
      int la = (v + 8 * hi) + 16 * p_s;
      p_sw[w * 512 + la * 16 + p_e0]     = (_Float16)p0;
      p_sw[w * 512 + la * 16 + p_e0 + 8] = (_Float16)p1;
    }
    __syncthreads();

    // O += P @ V  (K = 32 keys; N = 256 in 16 frags)
    v16h pa = lds_frag(p_sw + w * 512, lane);
#pragma unroll
    for (int f = 0; f < 16; ++f) {
      v16h bv = lds_frag(v_sw + f * 512, lane);
      oacc[f] = wmma_f16(pa, bv, oacc[f]);
    }
    __syncthreads();
  }

  // epilogue: rows contiguous in p -> v8h stores
  _Float16* op = o + (size_t)b * DIM * HW;
#pragma unroll
  for (int f = 0; f < 16; ++f) {
    int c = 16 * f + nl;
    float inv;
    v8h hv;
#pragma unroll
    for (int v = 0; v < 8; ++v) {
      inv = 1.f / rsum[v];
      hv[v] = (_Float16)(oacc[f][v] * inv);
    }
    *(v8h*)(void*)(op + (size_t)c * HW + m0 + 8 * hi) = hv;
  }
}

// ---------------- orchestration ----------------
extern "C" void kernel_launch(void* const* d_in, const int* in_sizes, int n_in,
                              void* d_out, int out_size, void* d_ws, size_t ws_size,
                              hipStream_t stream) {
  (void)in_sizes; (void)n_in; (void)out_size; (void)ws_size;
  const float* x        = (const float*)d_in[0];
  const float* time_emb = (const float*)d_in[1];
  const float* gn1_s    = (const float*)d_in[2];
  const float* gn1_b    = (const float*)d_in[3];
  const float* conv1_w  = (const float*)d_in[4];
  const float* conv1_b  = (const float*)d_in[5];
  const float* mlp_w    = (const float*)d_in[6];
  const float* mlp_b    = (const float*)d_in[7];
  const float* gn2_s    = (const float*)d_in[8];
  const float* gn2_b    = (const float*)d_in[9];
  const float* conv2_w  = (const float*)d_in[10];
  const float* conv2_b  = (const float*)d_in[11];
  const float* an_s     = (const float*)d_in[12];
  const float* an_b     = (const float*)d_in[13];
  const float* qkv_w    = (const float*)d_in[14];
  const float* qkv_b    = (const float*)d_in[15];
  const float* out_w    = (const float*)d_in[16];
  const float* out_b    = (const float*)d_in[17];

  char* ws = (char*)d_ws;
  size_t off = 0;
  auto alloc = [&](size_t bytes) {
    void* p = ws + off;
    off += (bytes + 255) & ~(size_t)255;
    return p;
  };
  _Float16* w1h    = (_Float16*)alloc((size_t)9 * DIM * DIM * 2);
  _Float16* w2h    = (_Float16*)alloc((size_t)9 * DIM * DIM * 2);
  _Float16* wqkvh  = (_Float16*)alloc((size_t)DIM * 768 * 2);
  _Float16* wouth  = (_Float16*)alloc((size_t)DIM * DIM * 2);
  float*    temb   = (float*)   alloc((size_t)BATCH * DIM * 4);
  _Float16* h_half = (_Float16*)alloc((size_t)BATCH * DIM * HW * 2);
  float*    h1     = (float*)   alloc((size_t)BATCH * DIM * HW * 4);
  float*    xr     = (float*)   alloc((size_t)BATCH * DIM * HW * 4);
  _Float16* qkvh   = (_Float16*)alloc((size_t)BATCH * 768 * HW * 2);
  _Float16* oh     = (_Float16*)alloc((size_t)BATCH * DIM * HW * 2);

  pack_weights<<<5632, 256, 0, stream>>>(conv1_w, conv2_w, qkv_w, out_w,
                                         w1h, w2h, wqkvh, wouth);
  time_mlp<<<BATCH, 256, 0, stream>>>(time_emb, mlp_w, mlp_b, temb);
  gn_silu<<<BATCH * GNUM, 256, 0, stream>>>(x, gn1_s, gn1_b, h_half, 1);
  conv3x3_wmma<<<BATCH * 64 * 4, 256, 0, stream>>>(h_half, w1h, conv1_b,
                                                   temb, nullptr, h1);
  gn_silu<<<BATCH * GNUM, 256, 0, stream>>>(h1, gn2_s, gn2_b, h_half, 1);
  conv3x3_wmma<<<BATCH * 64 * 4, 256, 0, stream>>>(h_half, w2h, conv2_b,
                                                   nullptr, x, xr);
  gn_silu<<<BATCH * GNUM, 256, 0, stream>>>(xr, an_s, an_b, h_half, 0);
  gemm1x1_wmma<<<BATCH * 64 * 12, 256, 0, stream>>>(h_half, wqkvh, qkv_b,
                                                    nullptr, nullptr, qkvh, 768);
  attn_wmma<<<BATCH * 32, 256, 0, stream>>>(qkvh, oh);
  gemm1x1_wmma<<<BATCH * 64 * 4, 256, 0, stream>>>(oh, wouth, out_b,
                                                   xr, (float*)d_out, nullptr, 256);
}
